// MemModule_58858231824621
// MI455X (gfx1250) — matrix operand order repaired and checked by reading, not compiled, not run on hardware
//
#include <hip/hip_runtime.h>

typedef __bf16 bf16;
typedef __attribute__((ext_vector_type(16))) __bf16 v16bf;
typedef __attribute__((ext_vector_type(8)))  __bf16 v8bf;
typedef __attribute__((ext_vector_type(8)))  float  v8f;

#define THREADS 256

// ---------------- row-wise L2 normalize, fp32 -> bf16 (zero-fill pad rows) ----------------
__global__ __launch_bounds__(THREADS)
void rownorm_bf16_kernel(const float* __restrict__ src, bf16* __restrict__ dst,
                         int rowsReal, int K) {
  const int row = blockIdx.x;
  const int tid = threadIdx.x;
  bf16* d = dst + (size_t)row * K;
  if (row >= rowsReal) {
    for (int k = tid; k < K; k += THREADS) d[k] = (bf16)0.0f;
    return;
  }
  __shared__ float red[THREADS];
  const float* s = src + (size_t)row * K;
  float ss = 0.0f;
  for (int k = tid; k < K; k += THREADS) { float v = s[k]; ss += v * v; }
  red[tid] = ss; __syncthreads();
  for (int off = THREADS / 2; off > 0; off >>= 1) {
    if (tid < off) red[tid] += red[tid + off];
    __syncthreads();
  }
  const float inv = 1.0f / fmaxf(sqrtf(red[0]), 1e-12f);
  for (int k = tid; k < K; k += THREADS) d[k] = (bf16)(s[k] * inv);
}

// ---------------- plain fp32 -> bf16 row copy (zero-fill pad rows) ----------------
__global__ __launch_bounds__(THREADS)
void rowcopy_bf16_kernel(const float* __restrict__ src, bf16* __restrict__ dst,
                         int rowsReal, int K) {
  const int row = blockIdx.x;
  const int tid = threadIdx.x;
  bf16* d = dst + (size_t)row * K;
  if (row < rowsReal) {
    const float* s = src + (size_t)row * K;
    for (int k = tid; k < K; k += THREADS) d[k] = (bf16)s[k];
  } else {
    for (int k = tid; k < K; k += THREADS) d[k] = (bf16)0.0f;
  }
}

// ---------------- WMMA bf16 GEMM:  out[i,j] = sum_k A[i,k] * B[j,k] (+ bias[i]) ----------
// A: [rowsA_pad, K] bf16 row-major; B: [rowsB_pad, K] bf16 row-major.
// Block = 256 threads = 8 waves; block tile 32(M) x 256(N); wave tile 16 x 64.
// Per wave: 1 A-fragment reused across 4 v_wmma_f32_16x16x32_bf16 per K-step.
__global__ __launch_bounds__(THREADS)
void gemm_bf16_wmma_kernel(const bf16* __restrict__ A, const bf16* __restrict__ B,
                           float* __restrict__ outF, bf16* __restrict__ outB,
                           const float* __restrict__ bias,
                           int K, int nOut, int ldo) {
  const int lane = threadIdx.x & 31;
  const int wave = threadIdx.x >> 5;
  const int wm = wave >> 2;                       // 0..1
  const int wn = wave & 3;                        // 0..3
  const int m0 = blockIdx.x * 32 + wm * 16;
  const int n0 = blockIdx.y * 256 + wn * 64;

  // A fragment (16x32 bf16): lane<16 holds row M=lane, K=0..15 contiguous;
  //                          lane>=16 holds row M=lane-16, K=16..31.
  const int frow = lane & 15;
  const int akl  = (lane < 16) ? 0 : 16;
  // B fragment (32x16 bf16): lane<16 holds col N=lane, K=0..15 contiguous;
  //                          lane>=16 holds col N=lane-16, K=16..31.
  const int bkl  = (lane < 16) ? 0 : 16;

  const bf16* Ap  = A + (size_t)(m0 + frow) * K;
  const bf16* Bp0 = B + (size_t)(n0 +  0 + frow) * K + bkl;
  const bf16* Bp1 = B + (size_t)(n0 + 16 + frow) * K + bkl;
  const bf16* Bp2 = B + (size_t)(n0 + 32 + frow) * K + bkl;
  const bf16* Bp3 = B + (size_t)(n0 + 48 + frow) * K + bkl;

  v8f acc0 = {}, acc1 = {}, acc2 = {}, acc3 = {};

  for (int k0 = 0; k0 < K; k0 += 32) {
    v8bf alo = *(const v8bf*)(Ap + k0 + akl);
    v8bf ahi = *(const v8bf*)(Ap + k0 + 8 + akl);
    v16bf a = __builtin_shufflevector(alo, ahi,
        0, 1, 2, 3, 4, 5, 6, 7, 8, 9, 10, 11, 12, 13, 14, 15);
    v16bf b0 = *(const v16bf*)(Bp0 + k0);
    v16bf b1 = *(const v16bf*)(Bp1 + k0);
    v16bf b2 = *(const v16bf*)(Bp2 + k0);
    v16bf b3 = *(const v16bf*)(Bp3 + k0);
    acc0 = __builtin_amdgcn_wmma_f32_16x16x32_bf16(false, a, false, b0, (short)0, acc0, false, false);
    acc1 = __builtin_amdgcn_wmma_f32_16x16x32_bf16(false, a, false, b1, (short)0, acc1, false, false);
    acc2 = __builtin_amdgcn_wmma_f32_16x16x32_bf16(false, a, false, b2, (short)0, acc2, false, false);
    acc3 = __builtin_amdgcn_wmma_f32_16x16x32_bf16(false, a, false, b3, (short)0, acc3, false, false);
  }

  // C/D layout: VGPR r -> M = m0 + r + (lane>=16 ? 8 : 0), N = n0 + 16*j + (lane&15)
  const int colc  = lane & 15;
  const int rbase = m0 + ((lane >= 16) ? 8 : 0);
#pragma unroll
  for (int r = 0; r < 8; ++r) {
    const int row = rbase + r;
    const float brow = bias ? bias[row] : 0.0f;
    const float vals[4] = {acc0[r], acc1[r], acc2[r], acc3[r]};
#pragma unroll
    for (int j = 0; j < 4; ++j) {
      const int col = n0 + 16 * j + colc;
      if (col < nOut) {
        const float v = vals[j] + brow;
        const size_t o = (size_t)row * ldo + col;
        if (outF) outF[o] = v;
        if (outB) outB[o] = (bf16)v;
      }
    }
  }
}

// ---------------- fused softmax + top-20 threshold + L2 renorm + att + entropy ------------
#define MPAD 2048
#define TOPK 20
__global__ __launch_bounds__(THREADS)
void softmax_topk_kernel(const float* __restrict__ logits,
                         float* __restrict__ att,      // [B, Mreal, N]
                         bf16*  __restrict__ alphaB,   // [T, MPAD]
                         float* __restrict__ entRow,   // [T]
                         int Mreal, int N) {
  __shared__ float ex[MPAD];    // mutable copy for extraction
  __shared__ float orig[MPAD];  // pristine exp values
  __shared__ float redv[THREADS];
  __shared__ int   redi[THREADS];
  __shared__ float sh_thr;

  const int t = blockIdx.x;
  const int tid = threadIdx.x;
  const float* lrow = logits + (size_t)t * MPAD;

  // 1) row max
  float lv[8];
  float lmax = -3.4e38f;
#pragma unroll
  for (int j = 0; j < 8; ++j) {
    const int m = tid + j * THREADS;
    const float v = (m < Mreal) ? lrow[m] : -3.4e38f;
    lv[j] = v;
    lmax = fmaxf(lmax, v);
  }
  redv[tid] = lmax; __syncthreads();
  for (int off = THREADS / 2; off > 0; off >>= 1) {
    if (tid < off) redv[tid] = fmaxf(redv[tid], redv[tid + off]);
    __syncthreads();
  }
  const float mx = redv[0]; __syncthreads();

  // 2) exp + sum
  float lsum = 0.0f;
#pragma unroll
  for (int j = 0; j < 8; ++j) {
    const int m = tid + j * THREADS;
    const float e = (m < Mreal) ? __expf(lv[j] - mx) : 0.0f;
    ex[m] = e; orig[m] = e;
    lsum += e;
  }
  redv[tid] = lsum; __syncthreads();
  for (int off = THREADS / 2; off > 0; off >>= 1) {
    if (tid < off) redv[tid] += redv[tid + off];
    __syncthreads();
  }
  const float sum = redv[0]; __syncthreads();

  // 3) 20 successive argmax extractions -> exact 20th-largest (exp domain,
  //    monotone with alpha so the threshold semantics match the reference)
  for (int it = 0; it < TOPK; ++it) {
    float bv = -1.0f; int bi = 0;
#pragma unroll
    for (int j = 0; j < 8; ++j) {
      const int m = tid + j * THREADS;
      const float v = ex[m];
      if (v > bv) { bv = v; bi = m; }
    }
    redv[tid] = bv; redi[tid] = bi; __syncthreads();
    for (int off = THREADS / 2; off > 0; off >>= 1) {
      if (tid < off && redv[tid + off] > redv[tid]) {
        redv[tid] = redv[tid + off]; redi[tid] = redi[tid + off];
      }
      __syncthreads();
    }
    if (tid == 0) { sh_thr = redv[0]; ex[redi[0]] = -1.0f; }
    __syncthreads();
  }
  const float thr = sh_thr;   // 20th-largest exp value

  // 4) mask (alpha >= thr  <=>  exp >= thr_exp) + L2 renorm
  float av[8];
  float ssq = 0.0f;
#pragma unroll
  for (int j = 0; j < 8; ++j) {
    const int m = tid + j * THREADS;
    const float a = (orig[m] >= thr) ? (orig[m] / sum) : 0.0f;
    av[j] = a; ssq += a * a;
  }
  redv[tid] = ssq; __syncthreads();
  for (int off = THREADS / 2; off > 0; off >>= 1) {
    if (tid < off) redv[tid] += redv[tid + off];
    __syncthreads();
  }
  const float invn = 1.0f / fmaxf(sqrtf(redv[0]), 1e-12f);
  __syncthreads();

  // 5) outputs: transposed att, bf16 alpha row, per-row entropy
  const int b = t / N, n = t % N;
  bf16* ab = alphaB + (size_t)t * MPAD;
  float ent = 0.0f;
#pragma unroll
  for (int j = 0; j < 8; ++j) {
    const int m = tid + j * THREADS;
    const float an = av[j] * invn;
    if (m < Mreal) {
      att[(size_t)b * Mreal * N + (size_t)m * N + n] = an;
      ent -= an * __logf(an + 1e-12f);
    }
    ab[m] = (bf16)an;   // pad lanes write exact zero
  }
  redv[tid] = ent; __syncthreads();
  for (int off = THREADS / 2; off > 0; off >>= 1) {
    if (tid < off) redv[tid] += redv[tid + off];
    __syncthreads();
  }
  if (tid == 0) entRow[t] = redv[0];
}

// ---------------- MSE block partials (deterministic) ----------------
__global__ __launch_bounds__(THREADS)
void mse_partial_kernel(const float* __restrict__ x, const float* __restrict__ y,
                        float* __restrict__ partials) {
  __shared__ float red[THREADS];
  const int tid = threadIdx.x;
  float s = 0.0f;
#pragma unroll
  for (int j = 0; j < 4; ++j) {
    const size_t i = (size_t)blockIdx.x * 1024 + tid + j * THREADS;
    const float d = x[i] - y[i];
    s += d * d;
  }
  red[tid] = s; __syncthreads();
  for (int off = THREADS / 2; off > 0; off >>= 1) {
    if (tid < off) red[tid] += red[tid + off];
    __syncthreads();
  }
  if (tid == 0) partials[blockIdx.x] = red[0];
}

// ---------------- final loss = mse_mean + ent_mean * w ----------------
__global__ __launch_bounds__(THREADS)
void final_loss_kernel(const float* __restrict__ msePart, int nP,
                       const float* __restrict__ entRow, int T,
                       float* __restrict__ lossOut, float invTC, float entW) {
  __shared__ float r1[THREADS], r2[THREADS];
  const int tid = threadIdx.x;
  float s1 = 0.0f, s2 = 0.0f;
  for (int i = tid; i < nP; i += THREADS) s1 += msePart[i];
  for (int i = tid; i < T;  i += THREADS) s2 += entRow[i];
  r1[tid] = s1; r2[tid] = s2; __syncthreads();
  for (int off = THREADS / 2; off > 0; off >>= 1) {
    if (tid < off) { r1[tid] += r1[tid + off]; r2[tid] += r2[tid + off]; }
    __syncthreads();
  }
  if (tid == 0) lossOut[0] = r1[0] * invTC + (r2[0] / (float)T) * entW;
}

// =========================================================================================
extern "C" void kernel_launch(void* const* d_in, const int* in_sizes, int n_in,
                              void* d_out, int out_size, void* d_ws, size_t ws_size,
                              hipStream_t stream) {
  (void)in_sizes; (void)n_in; (void)out_size; (void)ws_size;
  const float* x   = (const float*)d_in[0];   // [8, 2048, 1024]
  const float* mem = (const float*)d_in[1];   // [2000, 1024]
  const float* vw  = (const float*)d_in[2];   // [1024, 1024]
  const float* vb  = (const float*)d_in[3];   // [1024]
  float* out = (float*)d_out;

  constexpr int B = 8, N = 2048, C = 1024, M = 2000, Mp = 2048;
  constexpr int T = B * N;                    // 16384

  // workspace layout (all offsets 256B-aligned)
  char* ws = (char*)d_ws;
  bf16*  fn     = (bf16*)(ws + 0);            // [T,  C]  normalized feats      (32 MB)
  bf16*  wn     = (bf16*)(ws + 33554432ull);  // [Mp, C]  normalized memory     ( 4 MB)
  bf16*  memb   = (bf16*)(ws + 37748736ull);  // [Mp, C]  raw memory bf16       ( 4 MB)
  bf16*  vwb    = (bf16*)(ws + 41943040ull);  // [C,  C]  v_w bf16              ( 2 MB)
  bf16*  vmtb   = (bf16*)(ws + 44040192ull);  // [C,  Mp] vmem^T bf16           ( 4 MB)
  float* logits = (float*)(ws + 48234496ull); // [T,  Mp] fp32                  (128 MB)
  bf16*  alphab = (bf16*)(ws + 182452224ull); // [T,  Mp] masked/renormed alpha ( 64 MB)
  float* entRow = (float*)(ws + 249561088ull);// [T]
  float* msePt  = (float*)(ws + 249626624ull);// [T]

  float* y_out    = out;                                  // [T, C]
  float* att_out  = out + (size_t)T * C;                  // [B, M, N]
  float* loss_out = out + (size_t)T * C + (size_t)B * M * N;

  // 1) normalize / convert inputs to bf16 (zero-padded to Mp rows)
  rownorm_bf16_kernel<<<T,  THREADS, 0, stream>>>(x,   fn,   T, C);
  rownorm_bf16_kernel<<<Mp, THREADS, 0, stream>>>(mem, wn,   M, C);
  rowcopy_bf16_kernel<<<Mp, THREADS, 0, stream>>>(mem, memb, M, C);
  rowcopy_bf16_kernel<<<C,  THREADS, 0, stream>>>(vw,  vwb,  C, C);

  // 2) vmem^T[c,m] = sum_k v_w[c,k]*mem[m,k] + v_b[c]   (bf16 out, L2-resident 4 MB)
  gemm_bf16_wmma_kernel<<<dim3(C / 32, Mp / 256), THREADS, 0, stream>>>(
      vwb, memb, nullptr, vmtb, vb, C, Mp, Mp);

  // 3) logits[t,m] = sum_k fn[t,k]*wn[m,k]
  gemm_bf16_wmma_kernel<<<dim3(T / 32, Mp / 256), THREADS, 0, stream>>>(
      fn, wn, logits, nullptr, nullptr, C, Mp, Mp);

  // 4) softmax + exact top-20 threshold + L2 renorm + att + entropy
  softmax_topk_kernel<<<T, THREADS, 0, stream>>>(logits, att_out, alphab, entRow, M, N);

  // 5) y[t,c] = sum_m alpha[t,m] * vmem^T[c,m]   (K = Mp, alpha pad cols are zero)
  gemm_bf16_wmma_kernel<<<dim3(T / 32, C / 256), THREADS, 0, stream>>>(
      alphab, vmtb, y_out, nullptr, nullptr, Mp, C, C);

  // 6) loss = mean((x-y)^2) + mean(ent)*2e-4
  mse_partial_kernel<<<T, THREADS, 0, stream>>>(x, y_out, msePt);
  final_loss_kernel<<<1, THREADS, 0, stream>>>(msePt, T, entRow, T, loss_out,
                                               1.0f / ((float)T * (float)C), 0.0002f);
}
